// Decoder_39831526703230
// MI455X (gfx1250) — compile-verified, matrix operand order checked by reading, not run on hardware
//
#include <hip/hip_runtime.h>
#include <hip/hip_bf16.h>
#include <math.h>

typedef __attribute__((ext_vector_type(16))) __bf16 v16bf;
typedef __attribute__((ext_vector_type(8)))  float  v8f;

#define BB   2048
#define CC   64
#define DHD  512
#define CTXD 1536
static constexpr float SCALE = 0.044194173824159216f; // 1/sqrt(512)

__device__ __forceinline__ v8f zero8() {
    v8f r;
#pragma unroll
    for (int i = 0; i < 8; ++i) r[i] = 0.f;
    return r;
}

// ---------------------------------------------------------------------------
// Prep: pack context = [H_X | Ht_S | Eq_Q] to bf16, convert Wq/Wv to bf16,
// transpose Wk -> WkT (bf16) so WMMA B-fragments are contiguous loads.
// ---------------------------------------------------------------------------
__global__ __launch_bounds__(256) void prep_kernel(
    const float* __restrict__ Wq, const float* __restrict__ Wk, const float* __restrict__ Wv,
    const float* __restrict__ HX, const float* __restrict__ HtS, const float* __restrict__ EqQ,
    __bf16* __restrict__ ctx, __bf16* __restrict__ WqH,
    __bf16* __restrict__ WkT, __bf16* __restrict__ WvH)
{
    long i = (long)blockIdx.x * blockDim.x + threadIdx.x;
    const long N_ctx = (long)BB * CTXD;          // 3,145,728
    const long N_wq  = (long)DHD * CTXD;         //   786,432
    const long N_w   = (long)DHD * DHD;          //   262,144
    if (i < N_ctx) {
        long b = i / CTXD; int j = (int)(i % CTXD);
        float v = (j < DHD)   ? HX[b * DHD + j]
                : (j < 2*DHD) ? HtS[b * DHD + (j - DHD)]
                              : EqQ[b * DHD + (j - 2*DHD)];
        ctx[i] = (__bf16)v;
    } else if (i < N_ctx + N_wq) {
        long k = i - N_ctx;
        WqH[k] = (__bf16)Wq[k];
    } else if (i < N_ctx + N_wq + N_w) {
        long k = i - N_ctx - N_wq;
        int d = (int)(k / DHD), e = (int)(k % DHD);
        WkT[k] = (__bf16)Wk[(long)e * DHD + d];   // WkT[d][e] = Wk[e][d]
    } else if (i < N_ctx + N_wq + 2 * N_w) {
        long k = i - N_ctx - N_wq - N_w;
        WvH[k] = (__bf16)Wv[k];
    }
}

// ---------------------------------------------------------------------------
// Batched GEMM via v_wmma_f32_16x16x32_bf16.
//   out[m][n] = sum_k A[m][k] * Bm[n][k]       (Bm row n = WMMA B column n)
// Per block: 64 rows x 128 cols; wave w owns one 16-col strip, 4 m-tiles.
// ---------------------------------------------------------------------------
__global__ __launch_bounds__(256) void gemm_wmma_kernel(
    const __bf16* __restrict__ A, int lda,
    const __bf16* __restrict__ Bm, int ldb,
    float* __restrict__ outF, __bf16* __restrict__ outH, int ldc,
    int ksteps)
{
    const int lane = threadIdx.x & 31;
    const int wave = threadIdx.x >> 5;
    const int m0 = blockIdx.x * 64;
    const int n0 = blockIdx.y * 128 + wave * 16;
    const int r  = lane & 15;             // row-within-fragment
    const int kh = (lane >> 4) << 4;      // +16 k-offset for high half-wave

    v8f acc[4];
#pragma unroll
    for (int t = 0; t < 4; ++t) acc[t] = zero8();

    const __bf16* bp = Bm + (long)(n0 + r) * ldb + kh;
    const __bf16* ap = A  + (long)(m0 + r) * lda + kh;

    for (int ks = 0; ks < ksteps; ++ks) {
        v16bf bf = *(const v16bf*)(bp + ks * 32);
#pragma unroll
        for (int mt = 0; mt < 4; ++mt) {
            v16bf af = *(const v16bf*)(ap + (long)mt * 16 * lda + ks * 32);
            acc[mt] = __builtin_amdgcn_wmma_f32_16x16x32_bf16(
                false, af, false, bf, (short)0, acc[mt], false, false);
        }
    }

    const int col = n0 + r;
    const int rhi = (lane >> 4) * 8;      // M offset for high half-wave
#pragma unroll
    for (int mt = 0; mt < 4; ++mt) {
#pragma unroll
        for (int v = 0; v < 8; ++v) {
            int row = m0 + mt * 16 + rhi + v;
            float val = acc[mt][v];
            if (outF) outF[(long)row * ldc + col] = val;
            if (outH) outH[(long)row * ldc + col] = (__bf16)val;
        }
    }
}

// ---------------------------------------------------------------------------
// attn1: per batch element b:
//   G = Ht_C[b] + cap_emb[caps] + dist_emb[dists]  (staged to LDS, bf16)
//   logits = scale * G @ qbar[b];  w = softmax(logits);  gbar[b] = w^T G
// ---------------------------------------------------------------------------
#define GPITCH 514
__global__ __launch_bounds__(256) void attn1_kernel(
    const float* __restrict__ HtC, const int* __restrict__ caps, const int* __restrict__ dists,
    const float* __restrict__ cap_emb, const float* __restrict__ dist_emb,
    const float* __restrict__ qbar, __bf16* __restrict__ gbar)
{
    const int b = blockIdx.x;
    const int tid = threadIdx.x;
    __shared__ __bf16 Gs[CC * GPITCH];
    __shared__ float qb[DHD];
    __shared__ float part[4][CC];
    __shared__ float lg[CC];
    __shared__ float wgt[CC];

    for (int i = tid * 4; i < CC * DHD; i += 256 * 4) {
        int rr = i >> 9; int d = i & 511;
        float4 h  = *(const float4*)(HtC + ((long)b * CC + rr) * DHD + d);
        int cp = caps[b * CC + rr]; int dz = dists[b * CC + rr];
        float4 ce = *(const float4*)(cap_emb + (long)cp * DHD + d);
        float4 de = *(const float4*)(dist_emb + (long)dz * DHD + d);
        __bf16* g = &Gs[rr * GPITCH + d];
        g[0] = (__bf16)(h.x + ce.x + de.x); g[1] = (__bf16)(h.y + ce.y + de.y);
        g[2] = (__bf16)(h.z + ce.z + de.z); g[3] = (__bf16)(h.w + ce.w + de.w);
    }
    qb[tid]       = qbar[(long)b * DHD + tid];
    qb[tid + 256] = qbar[(long)b * DHD + tid + 256];
    __syncthreads();

    { // logits partial dots: 4 threads per core row
        int c = tid & 63, q = tid >> 6;
        const __bf16* grow = &Gs[c * GPITCH + q * 128];
        const float* qp = &qb[q * 128];
        float s = 0.f;
#pragma unroll 8
        for (int d = 0; d < 128; ++d) s += (float)grow[d] * qp[d];
        part[q][c] = s;
    }
    __syncthreads();
    if (tid < CC)
        lg[tid] = (part[0][tid] + part[1][tid] + part[2][tid] + part[3][tid]) * SCALE;
    __syncthreads();
    if (tid < CC) {
        float m = -INFINITY;
        for (int i = 0; i < CC; ++i) m = fmaxf(m, lg[i]);
        float ssum = 0.f;
        for (int i = 0; i < CC; ++i) ssum += __expf(lg[i] - m);
        wgt[tid] = __expf(lg[tid] - m) / ssum;
    }
    __syncthreads();
    { // gbar[d] = sum_c w[c] * G[c][d]; thread t -> d = 2t, 2t+1
        float a0 = 0.f, a1 = 0.f;
        int d = tid * 2;
#pragma unroll 4
        for (int c = 0; c < CC; ++c) {
            float w = wgt[c];
            a0 += w * (float)Gs[c * GPITCH + d];
            a1 += w * (float)Gs[c * GPITCH + d + 1];
        }
        gbar[(long)b * DHD + d]     = (__bf16)a0;
        gbar[(long)b * DHD + d + 1] = (__bf16)a1;
    }
}

// ---------------------------------------------------------------------------
// attn2: per batch element b:
//   u = scale * G @ kbar[b]; mask (cap < need -> -inf); out = softmax(u)
// ---------------------------------------------------------------------------
__global__ __launch_bounds__(256) void attn2_kernel(
    const float* __restrict__ HtC, const int* __restrict__ caps, const int* __restrict__ dists,
    const float* __restrict__ cap_emb, const float* __restrict__ dist_emb,
    const float* __restrict__ kbar, const unsigned char* __restrict__ dbl,
    float* __restrict__ out)
{
    const int b = blockIdx.x;
    const int tid = threadIdx.x;
    __shared__ __bf16 Gs[CC * GPITCH];
    __shared__ float kb[DHD];
    __shared__ float part[4][CC];
    __shared__ float lg[CC];

    for (int i = tid * 4; i < CC * DHD; i += 256 * 4) {
        int rr = i >> 9; int d = i & 511;
        float4 h  = *(const float4*)(HtC + ((long)b * CC + rr) * DHD + d);
        int cp = caps[b * CC + rr]; int dz = dists[b * CC + rr];
        float4 ce = *(const float4*)(cap_emb + (long)cp * DHD + d);
        float4 de = *(const float4*)(dist_emb + (long)dz * DHD + d);
        __bf16* g = &Gs[rr * GPITCH + d];
        g[0] = (__bf16)(h.x + ce.x + de.x); g[1] = (__bf16)(h.y + ce.y + de.y);
        g[2] = (__bf16)(h.z + ce.z + de.z); g[3] = (__bf16)(h.w + ce.w + de.w);
    }
    kb[tid]       = kbar[(long)b * DHD + tid];
    kb[tid + 256] = kbar[(long)b * DHD + tid + 256];
    __syncthreads();

    {
        int c = tid & 63, q = tid >> 6;
        const __bf16* grow = &Gs[c * GPITCH + q * 128];
        const float* kp = &kb[q * 128];
        float s = 0.f;
#pragma unroll 8
        for (int d = 0; d < 128; ++d) s += (float)grow[d] * kp[d];
        part[q][c] = s;
    }
    __syncthreads();
    if (tid < CC) {
        float u = (part[0][tid] + part[1][tid] + part[2][tid] + part[3][tid]) * SCALE;
        int need = dbl[b] ? 2 : 1;
        if (caps[b * CC + tid] < need) u = -INFINITY;
        lg[tid] = u;
    }
    __syncthreads();
    if (tid < CC) {
        float m = -INFINITY;
        for (int i = 0; i < CC; ++i) m = fmaxf(m, lg[i]);
        float ssum = 0.f;
        for (int i = 0; i < CC; ++i) ssum += __expf(lg[i] - m);
        out[(long)b * CC + tid] = __expf(lg[tid] - m) / ssum;
    }
}

// ---------------------------------------------------------------------------
extern "C" void kernel_launch(void* const* d_in, const int* in_sizes, int n_in,
                              void* d_out, int out_size, void* d_ws, size_t ws_size,
                              hipStream_t stream) {
    const float* HtC       = (const float*)d_in[0];
    const int*   caps      = (const int*)d_in[1];
    const int*   dists     = (const int*)d_in[2];
    const float* HX        = (const float*)d_in[3];
    const float* HtS       = (const float*)d_in[4];
    const float* EqQ       = (const float*)d_in[5];
    const unsigned char* dbl = (const unsigned char*)d_in[6];   // jnp bool = 1 byte
    const float* cap_emb   = (const float*)d_in[7];
    const float* dist_emb  = (const float*)d_in[8];
    const float* Wq        = (const float*)d_in[9];
    const float* Wk        = (const float*)d_in[10];
    const float* Wv        = (const float*)d_in[11];
    float* out = (float*)d_out;

    char* ws = (char*)d_ws;
    __bf16* ctx  = (__bf16*)(ws + 0);          // 2048*1536*2 = 6,291,456
    __bf16* WqH  = (__bf16*)(ws + 6291456);    //  512*1536*2 = 1,572,864
    __bf16* WkT  = (__bf16*)(ws + 7864320);    //   512*512*2 =   524,288
    __bf16* WvH  = (__bf16*)(ws + 8388608);    //   512*512*2 =   524,288
    __bf16* Qh   = (__bf16*)(ws + 8912896);    //  2048*512*2 = 2,097,152
    float*  qbar = (float* )(ws + 11010048);   //  2048*512*4 = 4,194,304
    __bf16* gbar = (__bf16*)(ws + 15204352);   //  2048*512*2 = 2,097,152
    __bf16* gli  = (__bf16*)(ws + 17301504);   //  2048*512*2 = 2,097,152
    float*  kbar = (float* )(ws + 19398656);   //  2048*512*4 = 4,194,304

    // 1) pack/convert inputs to bf16 (4,456,448 elements -> 17408 blocks)
    prep_kernel<<<17408, 256, 0, stream>>>(Wq, Wk, Wv, HX, HtS, EqQ, ctx, WqH, WkT, WvH);

    // 2) Q = ctx @ Wq^T      [2048 x 512], Kdim = 1536
    gemm_wmma_kernel<<<dim3(32, 4), 256, 0, stream>>>(ctx, CTXD, WqH, CTXD, nullptr, Qh, DHD, 48);

    // 3) qbar = Q @ Wk       [2048 x 512], Kdim = 512 (B-frags = WkT rows)
    gemm_wmma_kernel<<<dim3(32, 4), 256, 0, stream>>>(Qh, DHD, WkT, DHD, qbar, nullptr, DHD, 16);

    // 4) logits/softmax/gbar per batch element
    attn1_kernel<<<BB, 256, 0, stream>>>(HtC, caps, dists, cap_emb, dist_emb, qbar, gbar);

    // 5) glimpse = gbar @ Wv^T
    gemm_wmma_kernel<<<dim3(32, 4), 256, 0, stream>>>(gbar, DHD, WvH, DHD, nullptr, gli, DHD, 16);

    // 6) kbar = glimpse @ Wk
    gemm_wmma_kernel<<<dim3(32, 4), 256, 0, stream>>>(gli, DHD, WkT, DHD, kbar, nullptr, DHD, 16);

    // 7) u, capacity mask, final softmax
    attn2_kernel<<<BB, 256, 0, stream>>>(HtC, caps, dists, cap_emb, dist_emb, kbar, dbl, out);
}
